// PhyNEO_GNN_V2_27968827032295
// MI455X (gfx1250) — matrix operand order repaired and checked by reading, not compiled, not run on hardware
//
#include <hip/hip_runtime.h>

// ---------------------------------------------------------------------------
// PhyNEO GNN for MI455X (gfx1250, wave32, WMMA).
// bf16 WMMA (f32 accumulate). Weights pre-transposed to [N][K] bf16 (B side =
// two global_load_b128 per fragment). Activations keep f32 masters (residual /
// LayerNorm precision) plus bf16 mirrors so A fragments are pure b128 loads
// with no per-read conversion VALU. Conversions use native bf16 casts so the
// backend can emit packed cvt instructions.
// ---------------------------------------------------------------------------

#define HID   128
#define NN    10000
#define NE    160000
#define LAYERS 4
#define LN_EPS 1e-6f

typedef __bf16 bf16_t;
typedef bf16_t v16bf __attribute__((ext_vector_type(16)));
typedef bf16_t v8bf  __attribute__((ext_vector_type(8)));
typedef float  v8f   __attribute__((ext_vector_type(8)));

__device__ __forceinline__ bf16_t f2bf(float f) {
  return (bf16_t)f;                      // fptrunc, round-to-nearest-even
}

__device__ __forceinline__ void lds_fence() {
  // LDS ops are in-order within a wave on CDNA5; stop compiler code motion.
  asm volatile("" ::: "memory");
}

// A fragment (16x32 bf16, ISA 7.12.2 layout) from a bf16 row:
//   e=0..7  -> k = k0+e        (16 contiguous bytes at p)
//   e=8..15 -> k = k0+16+(e-8) (16 contiguous bytes at p+16)
__device__ __forceinline__ v16bf a_frag_bf(const bf16_t* __restrict__ p) {
  v8bf a0 = *(const v8bf*)p;
  v8bf a1 = *(const v8bf*)(p + 16);
  return __builtin_shufflevector(a0, a1, 0, 1, 2, 3, 4, 5, 6, 7,
                                 8, 9, 10, 11, 12, 13, 14, 15);
}

// Same, but from f32 spans (vector loads + packed converts).
__device__ __forceinline__ v16bf a_frag_f32(const float* __restrict__ p0,
                                            const float* __restrict__ p1) {
  v8bf a0 = __builtin_convertvector(*(const v8f*)p0, v8bf);
  v8bf a1 = __builtin_convertvector(*(const v8f*)p1, v8bf);
  return __builtin_shufflevector(a0, a1, 0, 1, 2, 3, 4, 5, 6, 7,
                                 8, 9, 10, 11, 12, 13, 14, 15);
}

// One K=32 WMMA step across all 8 column subtiles.
// Wt is bf16 column-major [128][K]; B fragment for lane L is two 16-byte spans
// at Wt[(nt*16+lm)*K + kBase + 8*half] and +16 elements (table order).
template <int K>
__device__ __forceinline__ void wmma_step(v16bf a, const bf16_t* __restrict__ Wt,
                                          int kBase, int lane, v8f* acc) {
  const int lm = lane & 15, half = lane >> 4;
  const bf16_t* base = Wt + (size_t)lm * K + kBase + (half << 3);
#pragma unroll
  for (int nt = 0; nt < 8; ++nt) {
    const v8bf* bp = (const v8bf*)(base + (size_t)nt * 16 * K);
    v8bf b0 = bp[0], b1 = bp[2];
    v16bf b = __builtin_shufflevector(b0, b1, 0, 1, 2, 3, 4, 5, 6, 7,
                                      8, 9, 10, 11, 12, 13, 14, 15);
    acc[nt] = __builtin_amdgcn_wmma_f32_16x16x32_bf16(
        false, a, false, b, (short)0, acc[nt], false, false);
  }
}

__device__ __forceinline__ void zero_acc(v8f* acc) {
#pragma unroll
  for (int nt = 0; nt < 8; ++nt)
#pragma unroll
    for (int r = 0; r < 8; ++r) acc[nt][r] = 0.f;
}

// ------------------------------- utility -----------------------------------
// src: f32 row-major [K][128]; dst: bf16 column-major [128][Kdst], zero-pad.
__global__ void transpose_cvt_kernel(const float* __restrict__ src,
                                     bf16_t* __restrict__ dst, int K, int Kdst) {
  int i = blockIdx.x * blockDim.x + threadIdx.x;
  if (i >= 128 * Kdst) return;
  int n = i / Kdst, k = i - n * Kdst;
  dst[i] = f2bf(k < K ? src[(size_t)k * 128 + n] : 0.f);
}

__global__ void zero_f32_kernel(float* __restrict__ p, int n) {
  int i = blockIdx.x * blockDim.x + threadIdx.x;
  if (i < n) p[i] = 0.f;
}

// --------------------------- embedding GEMMs -------------------------------
// nodes = node_features[N,64] @ Wn0 + bn0      (Wn0t: [128][64] bf16)
__global__ void embed_nodes_kernel(const float* __restrict__ nf,
                                   const bf16_t* __restrict__ Wt,
                                   const float* __restrict__ bias,
                                   float* __restrict__ nodes,
                                   bf16_t* __restrict__ nodes_bf) {
  int lane = threadIdx.x & 31, wid = threadIdx.x >> 5;
  int tile = blockIdx.x * 4 + wid;
  if (tile >= NN / 16) return;
  int lm = lane & 15, half = lane >> 4;
  int rowBase = tile * 16;
  const float* aRow = nf + (size_t)(rowBase + lm) * 64;
  v8f acc[8]; zero_acc(acc);
#pragma unroll
  for (int kt = 0; kt < 2; ++kt) {
    const float* p0 = aRow + kt * 32 + (half << 3);
    v16bf a = a_frag_f32(p0, p0 + 16);
    wmma_step<64>(a, Wt, kt * 32, lane, acc);
  }
#pragma unroll
  for (int nt = 0; nt < 8; ++nt)
#pragma unroll
    for (int r = 0; r < 8; ++r) {
      int col = nt * 16 + lm, M = r + (half << 3);
      float v = acc[nt][r] + bias[col];
      size_t idx = (size_t)(rowBase + M) * HID + col;
      nodes[idx] = v;
      nodes_bf[idx] = f2bf(v);
    }
}

// edges = edge_features[E,16] @ We0 + be0   (We0t: [128][32], rows k>=16 zero)
__global__ void embed_edges_kernel(const float* __restrict__ ef,
                                   const bf16_t* __restrict__ Wt,
                                   const float* __restrict__ bias,
                                   float* __restrict__ edges,
                                   bf16_t* __restrict__ edges_bf) {
  int lane = threadIdx.x & 31, wid = threadIdx.x >> 5;
  int tile = blockIdx.x * 4 + wid;
  if (tile >= NE / 16) return;
  int lm = lane & 15, half = lane >> 4;
  int rowBase = tile * 16;
  const float* aRow = ef + (size_t)(rowBase + lm) * 16;
  v8f acc[8]; zero_acc(acc);
  // K padded 16->32: e=0..7 -> k = 8*half + e (valid); e=8..15 -> k>=16 (zero)
  v8bf a0 = __builtin_convertvector(*(const v8f*)(aRow + (half << 3)), v8bf);
  v16bf a;
#pragma unroll
  for (int e = 0; e < 8; ++e) {
    a[e]     = a0[e];
    a[8 + e] = f2bf(0.f);
  }
  wmma_step<32>(a, Wt, 0, lane, acc);
#pragma unroll
  for (int nt = 0; nt < 8; ++nt)
#pragma unroll
    for (int r = 0; r < 8; ++r) {
      int col = nt * 16 + lm, M = r + (half << 3);
      float v = acc[nt][r] + bias[col];
      size_t idx = (size_t)(rowBase + M) * HID + col;
      edges[idx] = v;
      edges_bf[idx] = f2bf(v);
    }
}

// ----------------- fused edge update + message + aggregate ------------------
// new_edges = [edges|snd|rcv] @ We + be        (f32 to global, bf16 to LDS)
// msg       = [rcv|new_edges] @ Wm + bm        (never materialized)
// agg[rcv] += msg                              (atomic segment-sum)
__global__ void __launch_bounds__(128)
edge_msg_kernel(const bf16_t* __restrict__ nodes_bf,
                const bf16_t* __restrict__ edges_bf,
                const int* __restrict__ senders,
                const int* __restrict__ receivers,
                const bf16_t* __restrict__ Wet, const float* __restrict__ be,
                const bf16_t* __restrict__ Wmt, const float* __restrict__ bm,
                float* __restrict__ new_edges, float* __restrict__ agg) {
  __shared__ bf16_t tileS[4][16 * HID];
  int lane = threadIdx.x & 31, wid = threadIdx.x >> 5;
  int tile = blockIdx.x * 4 + wid;
  if (tile >= NE / 16) return;
  int lm = lane & 15, half = lane >> 4;
  int rowBase = tile * 16;
  int aRow = rowBase + lm;
  int snd = senders[aRow], rcv = receivers[aRow];
  const bf16_t* eRow = edges_bf + (size_t)aRow * HID;
  const bf16_t* sRow = nodes_bf + (size_t)snd * HID;
  const bf16_t* rRow = nodes_bf + (size_t)rcv * HID;
  bf16_t* Tw = &tileS[wid][0];

  v8f acc[8]; zero_acc(acc);
  // GEMM1: A = [edges | sender | receiver]  (16x384); each 32-wide kt block
  // lies within one 128-aligned source.
#pragma unroll
  for (int kt = 0; kt < 12; ++kt) {
    const bf16_t* src = (kt < 4) ? eRow : (kt < 8 ? sRow - 128 : rRow - 256);
    v16bf a = a_frag_bf(src + kt * 32 + (half << 3));
    wmma_step<384>(a, Wet, kt * 32, lane, acc);
  }

  // add bias; bf16 tile to LDS for GEMM2's K-side; f32 new_edges to global
#pragma unroll
  for (int nt = 0; nt < 8; ++nt)
#pragma unroll
    for (int r = 0; r < 8; ++r) {
      int col = nt * 16 + lm, M = r + (half << 3);
      float v = acc[nt][r] + be[col];
      Tw[M * HID + col] = f2bf(v);
      new_edges[(size_t)(rowBase + M) * HID + col] = v;
    }
  lds_fence();

  zero_acc(acc);
  // GEMM2: A = [receiver | new_edges] (16x256); kt 0..3 global bf16,
  // kt 4..7 from the bf16 LDS tile (ds_load_b128).
#pragma unroll
  for (int kt = 0; kt < 4; ++kt) {
    v16bf a = a_frag_bf(rRow + kt * 32 + (half << 3));
    wmma_step<256>(a, Wmt, kt * 32, lane, acc);
  }
  const bf16_t* TwRow = Tw + lm * HID;
#pragma unroll
  for (int kt = 4; kt < 8; ++kt) {
    v16bf a = a_frag_bf(TwRow + (kt - 4) * 32 + (half << 3));
    wmma_step<256>(a, Wmt, kt * 32, lane, acc);
  }

  int tgt[8];
#pragma unroll
  for (int r = 0; r < 8; ++r) tgt[r] = receivers[rowBase + r + (half << 3)];
#pragma unroll
  for (int nt = 0; nt < 8; ++nt)
#pragma unroll
    for (int r = 0; r < 8; ++r) {
      int col = nt * 16 + lm;
      atomicAdd(&agg[(size_t)tgt[r] * HID + col], acc[nt][r] + bm[col]);
    }
}

// -------------- node update GEMM + residual + LayerNorm (fused) -------------
__global__ void __launch_bounds__(128)
node_update_kernel(float* __restrict__ nodes, bf16_t* __restrict__ nodes_bf,
                   const float* __restrict__ agg,
                   const bf16_t* __restrict__ Wnt, const float* __restrict__ bn,
                   const float* __restrict__ ls, const float* __restrict__ lb) {
  __shared__ float tileS[4][16 * HID];
  __shared__ float stats[4][32];
  int lane = threadIdx.x & 31, wid = threadIdx.x >> 5;
  int tile = blockIdx.x * 4 + wid;
  if (tile >= NN / 16) return;            // wave-level exit; no block barriers
  int lm = lane & 15, half = lane >> 4;
  int rowBase = tile * 16;
  const bf16_t* nRow = nodes_bf + (size_t)(rowBase + lm) * HID;
  const float*  gRow = agg + (size_t)(rowBase + lm) * HID;
  float* Tw = &tileS[wid][0];

  v8f acc[8]; zero_acc(acc);
#pragma unroll
  for (int kt = 0; kt < 4; ++kt) {        // k < 128: nodes (bf16 mirror)
    v16bf a = a_frag_bf(nRow + kt * 32 + (half << 3));
    wmma_step<256>(a, Wnt, kt * 32, lane, acc);
  }
#pragma unroll
  for (int kt = 4; kt < 8; ++kt) {        // k >= 128: agg (f32, read once)
    const float* p0 = gRow + (kt - 4) * 32 + (half << 3);
    v16bf a = a_frag_f32(p0, p0 + 16);
    wmma_step<256>(a, Wnt, kt * 32, lane, acc);
  }

  // residual add, stage f32 tile in LDS for LayerNorm
#pragma unroll
  for (int nt = 0; nt < 8; ++nt)
#pragma unroll
    for (int r = 0; r < 8; ++r) {
      int col = nt * 16 + lm, M = r + (half << 3);
      Tw[M * HID + col] =
          acc[nt][r] + bn[col] + nodes[(size_t)(rowBase + M) * HID + col];
    }
  lds_fence();

  if (lane < 16) {                        // one lane per row: stats
    float s = 0.f;
#pragma unroll 8
    for (int c = 0; c < HID; ++c) s += Tw[lane * HID + c];
    float mu = s * (1.f / HID);
    float v = 0.f;
#pragma unroll 8
    for (int c = 0; c < HID; ++c) {
      float d = Tw[lane * HID + c] - mu;
      v += d * d;
    }
    stats[wid][lane] = mu;
    stats[wid][16 + lane] = rsqrtf(v * (1.f / HID) + LN_EPS);
  }
  lds_fence();

  for (int r = 0; r < 16; ++r) {
    float mu = stats[wid][r], rstd = stats[wid][16 + r];
    for (int c = lane; c < HID; c += 32) {
      float x = (Tw[r * HID + c] - mu) * rstd * ls[c] + lb[c];
      size_t idx = (size_t)(rowBase + r) * HID + c;
      nodes[idx] = x;
      nodes_bf[idx] = f2bf(x);
    }
  }
}

// ------------------- edge residual + LayerNorm (wave/row) -------------------
__global__ void edge_ln_kernel(const float* __restrict__ new_edges,
                               float* __restrict__ edges,
                               bf16_t* __restrict__ edges_bf,
                               const float* __restrict__ ls,
                               const float* __restrict__ lb) {
  int lane = threadIdx.x & 31, wid = threadIdx.x >> 5;
  int row = blockIdx.x * 8 + wid;
  if (row >= NE) return;
  size_t base = (size_t)row * HID;
  float x[4], s = 0.f;
#pragma unroll
  for (int i = 0; i < 4; ++i) {
    x[i] = new_edges[base + lane + 32 * i] + edges[base + lane + 32 * i];
    s += x[i];
  }
  for (int off = 16; off; off >>= 1) s += __shfl_xor(s, off, 32);
  float mu = s * (1.f / HID), v = 0.f;
#pragma unroll
  for (int i = 0; i < 4; ++i) {
    float d = x[i] - mu;
    v += d * d;
  }
  for (int off = 16; off; off >>= 1) v += __shfl_xor(v, off, 32);
  float rstd = rsqrtf(v * (1.f / HID) + LN_EPS);
#pragma unroll
  for (int i = 0; i < 4; ++i) {
    int c = lane + 32 * i;
    float y = (x[i] - mu) * rstd * ls[c] + lb[c];
    edges[base + c] = y;
    edges_bf[base + c] = f2bf(y);
  }
}

// ---------------------- FF head + physical scaling --------------------------
__global__ void ff_head_kernel(const float* __restrict__ nodes,
                               const float* __restrict__ Wf1, const float* __restrict__ bf1,
                               const float* __restrict__ Wf2, const float* __restrict__ bf2,
                               const float* __restrict__ Wf3, const float* __restrict__ bf3,
                               float* __restrict__ out) {
  __shared__ float sh[8][112];           // 64 h1 + 32 h2 + 6 raw per wave
  int lane = threadIdx.x & 31, wid = threadIdx.x >> 5;
  int node = blockIdx.x * 8 + wid;
  if (node >= NN) return;
  const float* x = nodes + (size_t)node * HID;
  float* s = sh[wid];
#pragma unroll
  for (int j0 = 0; j0 < 2; ++j0) {
    int j = lane + 32 * j0;
    float a = bf1[j];
    for (int k = 0; k < HID; ++k) a += x[k] * Wf1[k * 64 + j];
    s[j] = a / (1.f + expf(-a));         // silu
  }
  lds_fence();
  {
    int j = lane;
    float a = bf2[j];
    for (int k = 0; k < 64; ++k) a += s[k] * Wf2[k * 32 + j];
    s[64 + j] = a / (1.f + expf(-a));
  }
  lds_fence();
  if (lane < 6) {
    float a = bf3[lane];
    for (int k = 0; k < 32; ++k) a += s[64 + k] * Wf3[k * 6 + lane];
    s[96 + lane] = a;
  }
  lds_fence();
  if (lane == 0) {
    float kappa = 2.f / (1.f + expf(-s[96]));
    out[(size_t)node * 7 + 0] = kappa;
    out[(size_t)node * 7 + 6] = 35.f * powf(kappa, -1.f / 3.f);
  } else if (lane <= 5) {
    out[(size_t)node * 7 + lane] = 100.f * expf(s[96 + lane]);
  }
}

// ---------------------------------------------------------------------------
extern "C" void kernel_launch(void* const* d_in, const int* in_sizes, int n_in,
                              void* d_out, int out_size, void* d_ws, size_t ws_size,
                              hipStream_t stream) {
  (void)in_sizes; (void)n_in; (void)out_size; (void)ws_size;
  const float* node_features = (const float*)d_in[0];
  const float* edge_features = (const float*)d_in[1];
  const int*   senders       = (const int*)d_in[2];
  const int*   receivers     = (const int*)d_in[3];
  const float* Wn0 = (const float*)d_in[4];
  const float* bn0 = (const float*)d_in[5];
  const float* We0 = (const float*)d_in[6];
  const float* be0 = (const float*)d_in[7];
  const float* We  = (const float*)d_in[8];
  const float* be  = (const float*)d_in[9];
  const float* Wm  = (const float*)d_in[10];
  const float* bm  = (const float*)d_in[11];
  const float* Wn  = (const float*)d_in[12];
  const float* bn  = (const float*)d_in[13];
  const float* ln_n_s = (const float*)d_in[14];
  const float* ln_n_b = (const float*)d_in[15];
  const float* ln_e_s = (const float*)d_in[16];
  const float* ln_e_b = (const float*)d_in[17];
  const float* Wf1 = (const float*)d_in[18];
  const float* bf1 = (const float*)d_in[19];
  const float* Wf2 = (const float*)d_in[20];
  const float* bf2 = (const float*)d_in[21];
  const float* Wf3 = (const float*)d_in[22];
  const float* bf3 = (const float*)d_in[23];
  float* out = (float*)d_out;

  // workspace partition
  char* w = (char*)d_ws;
  auto alloc = [&](size_t bytes) -> void* {
    void* p = (void*)w;
    w += (bytes + 255) & ~(size_t)255;
    return p;
  };
  float*  nodes     = (float*)alloc((size_t)NN * HID * 4);
  bf16_t* nodes_bf  = (bf16_t*)alloc((size_t)NN * HID * 2);
  float*  agg       = (float*)alloc((size_t)NN * HID * 4);
  float*  edges     = (float*)alloc((size_t)NE * HID * 4);
  bf16_t* edges_bf  = (bf16_t*)alloc((size_t)NE * HID * 2);
  float*  new_edges = (float*)alloc((size_t)NE * HID * 4);
  bf16_t* Wn0t = (bf16_t*)alloc((size_t)128 * 64 * 2);
  bf16_t* We0t = (bf16_t*)alloc((size_t)128 * 32 * 2);   // K padded to 32
  bf16_t* Wet  = (bf16_t*)alloc((size_t)LAYERS * 128 * 384 * 2);
  bf16_t* Wmt  = (bf16_t*)alloc((size_t)LAYERS * 128 * 256 * 2);
  bf16_t* Wnt  = (bf16_t*)alloc((size_t)LAYERS * 128 * 256 * 2);

  // weight transpose+convert: [K][128] f32 -> [128][Kdst] bf16 (zero-padded)
  auto tc = [&](const float* src, bf16_t* dst, int K, int Kdst) {
    transpose_cvt_kernel<<<(128 * Kdst + 255) / 256, 256, 0, stream>>>(src, dst, K, Kdst);
  };
  tc(Wn0, Wn0t, 64, 64);
  tc(We0, We0t, 16, 32);
  for (int l = 0; l < LAYERS; ++l) {
    tc(We + (size_t)l * 384 * 128, Wet + (size_t)l * 128 * 384, 384, 384);
    tc(Wm + (size_t)l * 256 * 128, Wmt + (size_t)l * 128 * 256, 256, 256);
    tc(Wn + (size_t)l * 256 * 128, Wnt + (size_t)l * 128 * 256, 256, 256);
  }

  // embeddings
  embed_nodes_kernel<<<(NN / 16 + 3) / 4, 128, 0, stream>>>(
      node_features, Wn0t, bn0, nodes, nodes_bf);
  embed_edges_kernel<<<NE / 16 / 4, 128, 0, stream>>>(
      edge_features, We0t, be0, edges, edges_bf);

  for (int l = 0; l < LAYERS; ++l) {
    zero_f32_kernel<<<(NN * HID + 255) / 256, 256, 0, stream>>>(agg, NN * HID);
    edge_msg_kernel<<<NE / 16 / 4, 128, 0, stream>>>(
        nodes_bf, edges_bf, senders, receivers,
        Wet + (size_t)l * 128 * 384, be + l * HID,
        Wmt + (size_t)l * 128 * 256, bm + l * HID,
        new_edges, agg);
    node_update_kernel<<<(NN / 16 + 3) / 4, 128, 0, stream>>>(
        nodes, nodes_bf, agg, Wnt + (size_t)l * 128 * 256, bn + l * HID,
        ln_n_s + l * HID, ln_n_b + l * HID);
    edge_ln_kernel<<<NE / 8, 256, 0, stream>>>(
        new_edges, edges, edges_bf, ln_e_s + l * HID, ln_e_b + l * HID);
  }

  ff_head_kernel<<<(NN + 7) / 8, 256, 0, stream>>>(
      nodes, Wf1, bf1, Wf2, bf2, Wf3, bf3, out);
}
